// MoE_730144440513
// MI455X (gfx1250) — matrix operand order repaired and checked by reading, not compiled, not run on hardware
//
#include <hip/hip_runtime.h>

#define H_DIM   1024
#define NEXP    16
#define I_DIM   128
#define T_TOK   16384            // B*S
#define TILE_M  64               // four 16-row WMMA tiles per block
#define NTILES  (TILE_M / 16)
#define MAX_TILES (T_TOK / TILE_M)

// LDS row strides in halves; multiples of 8 (16B-aligned rows for ds_load_b128),
// row-stride/16B odd -> wave's 32x16B A-frag reads spread evenly over banks.
#define XS_STRIDE 1048   // 1048*2/16 = 131 (odd)
#define AS_STRIDE 152    // 152*2/16  = 19  (odd)

#define FRAGS_PER_EXPERT 256     // (Nout/16) * (K/32) for both Wg and Wu
#define FRAG_HALVES      512     // 32 lanes x 16 halves

typedef __attribute__((ext_vector_type(16))) _Float16 v16h;
typedef __attribute__((ext_vector_type(8)))  _Float16 v8h;
typedef __attribute__((ext_vector_type(4)))  _Float16 h4;
typedef __attribute__((ext_vector_type(8)))  float    v8f;
typedef __attribute__((ext_vector_type(4)))  float    f4;

// A-fragment: per-lane K runs [koff, koff+8) and [koff+16, koff+24) are contiguous
// halves -> two 16B LDS loads, merged into one v16h by pure register placement.
__device__ __forceinline__ v16h load_afrag(const _Float16* p) {
    v8h lo = *(const v8h*)p;
    v8h hi = *(const v8h*)(p + 16);
    return __builtin_shufflevector(lo, hi, 0, 1, 2, 3, 4, 5, 6, 7,
                                   8, 9, 10, 11, 12, 13, 14, 15);
}

// ---------------------------------------------------------------- zero
__global__ __launch_bounds__(256) void moe_zero_kernel(float* __restrict__ out,
                                                       int* __restrict__ counts,
                                                       int n4) {
    int i = blockIdx.x * 256 + threadIdx.x;
    f4 z = {0.f, 0.f, 0.f, 0.f};
    if (i < n4) ((f4*)out)[i] = z;
    if (blockIdx.x == 0 && threadIdx.x < NEXP) counts[threadIdx.x] = 0;
}

// ---------------------------------------------------------------- weight pack
// frag(e, nt, kci): halves[lane*16 + s] = W[e][nt*16+lane%16][kci*32+k], k = 8*(lane/16)+(s&7)+16*(s>>3)
__global__ __launch_bounds__(256) void moe_pack_kernel(
    const float* __restrict__ Wg, const float* __restrict__ Wu,
    _Float16* __restrict__ pg, _Float16* __restrict__ pu) {
    __shared__ _Float16 sm[FRAG_HALVES];
    int fid = blockIdx.x;
    const float* src;  _Float16* dst;  int K, Nout;
    if (blockIdx.y == 0) { src = Wg; dst = pg; K = H_DIM; Nout = I_DIM; }
    else                 { src = Wu; dst = pu; K = I_DIM; Nout = H_DIM; }
    int e   = fid >> 8;
    int f   = fid & 255;
    int kch = K / 32;
    int nt  = f / kch;
    int kci = f % kch;
    size_t ebase = (size_t)e * Nout * K;
    int tid = threadIdx.x;
#pragma unroll
    for (int s = 0; s < 2; ++s) {
        int elem = tid + 256 * s;
        int j = elem >> 5, l = elem & 31;              // coalesced f32 reads over l
        float v = src[ebase + (size_t)(nt * 16 + j) * K + kci * 32 + l];
        sm[(j + 16 * ((l >> 3) & 1)) * 16 + (l & 7) + 8 * (l >> 4)] = (_Float16)v;  // transpose in LDS
    }
    __syncthreads();
    ((unsigned*)dst)[(size_t)fid * (FRAG_HALVES / 2) + tid] =
        ((const unsigned*)sm)[tid];
}

// ---------------------------------------------------------------- router
__global__ __launch_bounds__(256) void moe_router_kernel(
    const float* __restrict__ x, const float* __restrict__ Wr,
    int* __restrict__ counts, int* __restrict__ lists, float* __restrict__ gws) {
    __shared__ float sWr[NEXP * H_DIM];
    int tid = threadIdx.x;
    for (int i = tid; i < NEXP * H_DIM; i += 256) sWr[i] = Wr[i];
    __syncthreads();

    int lane = tid & 31;
    int wave = tid >> 5;
    int t = blockIdx.x * 8 + wave;
    const float* xr = x + (size_t)t * H_DIM;

    float acc[NEXP];
#pragma unroll
    for (int n = 0; n < NEXP; ++n) acc[n] = 0.f;
    for (int j = 0; j < H_DIM / 32; ++j) {
        float xv = xr[lane + 32 * j];
#pragma unroll
        for (int n = 0; n < NEXP; ++n)
            acc[n] = __builtin_fmaf(xv, sWr[n * H_DIM + lane + 32 * j], acc[n]);
    }
#pragma unroll
    for (int n = 0; n < NEXP; ++n) {
#pragma unroll
        for (int off = 16; off >= 1; off >>= 1)
            acc[n] += __shfl_xor(acc[n], off, 32);
    }
    if (lane == 0) {
        int i0 = 0; float v0 = acc[0];
#pragma unroll
        for (int n = 1; n < NEXP; ++n) if (acc[n] > v0) { v0 = acc[n]; i0 = n; }
        int i1 = -1; float v1 = -3.4e38f;
#pragma unroll
        for (int n = 0; n < NEXP; ++n)
            if (n != i0 && acc[n] > v1) { v1 = acc[n]; i1 = n; }
        float e1 = __expf(v1 - v0);
        float inv = 1.f / (1.f + e1);
        float w0 = inv, w1 = e1 * inv;
        int p0 = atomicAdd(&counts[i0], 1);
        lists[i0 * T_TOK + p0] = t; gws[i0 * T_TOK + p0] = w0;
        int p1 = atomicAdd(&counts[i1], 1);
        lists[i1 * T_TOK + p1] = t; gws[i1 * T_TOK + p1] = w1;
    }
}

// ---------------------------------------------------------------- expert MLP
__global__ __launch_bounds__(256) void moe_expert_kernel(
    const float* __restrict__ x,
    const _Float16* __restrict__ pg, const float* __restrict__ bg,
    const _Float16* __restrict__ pu, const float* __restrict__ bu,
    const int* __restrict__ counts, const int* __restrict__ lists,
    const float* __restrict__ gws, float* __restrict__ out) {
    int e = blockIdx.y;
    int tile = blockIdx.x;
    int cnt = counts[e];
    if (tile * TILE_M >= cnt) return;            // wave-uniform: EXEC all-1s below

    __shared__ _Float16 Xs[TILE_M * XS_STRIDE];  // ~131 KB
    __shared__ _Float16 As[TILE_M * AS_STRIDE];  // ~19 KB
    __shared__ int   toks[TILE_M];
    __shared__ float gwt[TILE_M];

    int tid = threadIdx.x;
    if (tid < TILE_M) {
        int r = tile * TILE_M + tid;
        int valid = r < cnt;
        int rr = valid ? r : tile * TILE_M;
        toks[tid] = lists[e * T_TOK + rr];
        gwt[tid]  = valid ? gws[e * T_TOK + r] : 0.f;   // zero gate kills padding
    }
    __syncthreads();

    // gather X tile -> f16 LDS; float4 global loads, 8B LDS stores
    for (int it = 0; it < (TILE_M * H_DIM / 4) / 256; ++it) {   // 64 iters
        int idx = tid + 256 * it;
        int row = idx >> 8;                  // 256 float4 per row
        int c4  = idx & 255;
        f4 v = ((const f4*)(x + (size_t)toks[row] * H_DIM))[c4];
        h4 q;
        q.x = (_Float16)v.x; q.y = (_Float16)v.y;
        q.z = (_Float16)v.z; q.w = (_Float16)v.w;
        *(h4*)&Xs[row * XS_STRIDE + c4 * 4] = q;
    }
    __syncthreads();

    int lane = tid & 31;
    int wave = tid >> 5;
    int m16  = lane & 15;
    int hi   = lane >> 4;
    int koff = hi * 8;

    // ---- GEMM1: wave w owns inner cols [16w,16w+16), all four 16-token tiles
    {
        v8f acc[NTILES] = {};
        const v16h* bp = (const v16h*)pg
                       + (size_t)(e * FRAGS_PER_EXPERT + wave * (H_DIM / 32)) * 32
                       + lane;
#pragma unroll 2
        for (int kci = 0; kci < H_DIM / 32; ++kci) {
            __builtin_prefetch((const void*)(bp + (kci + 6) * 32), 0, 1);
            v16h b = bp[kci * 32];
            int base = kci * 32 + koff;
#pragma unroll
            for (int t = 0; t < NTILES; ++t) {
                v16h a = load_afrag(&Xs[(16 * t + m16) * XS_STRIDE + base]);
                acc[t] = __builtin_amdgcn_wmma_f32_16x16x32_f16(
                    false, a, false, b, (short)0, acc[t], false, false);
            }
        }
        float bgv = bg[e * I_DIM + wave * 16 + m16];
#pragma unroll
        for (int t = 0; t < NTILES; ++t) {
#pragma unroll
            for (int g = 0; g < 8; ++g) {
                int m = 16 * t + g + 8 * hi;
                float u = acc[t][g] + bgv;
                As[m * AS_STRIDE + wave * 16 + m16] =
                    (_Float16)(u / (1.f + __expf(-u)));
            }
        }
    }
    __syncthreads();

    // ---- GEMM2: 64 output n-tiles of 16 over H, 8 per wave, all token tiles
    for (int j = 0; j < 8; ++j) {
        int nt = j * 8 + wave;
        int nb = nt * 16;
        v8f acc[NTILES] = {};
        const v16h* bp = (const v16h*)pu
                       + (size_t)(e * FRAGS_PER_EXPERT + nt * (I_DIM / 32)) * 32
                       + lane;
#pragma unroll
        for (int kci = 0; kci < I_DIM / 32; ++kci) {
            v16h b = bp[kci * 32];
            int base = kci * 32 + koff;
#pragma unroll
            for (int t = 0; t < NTILES; ++t) {
                v16h a = load_afrag(&As[(16 * t + m16) * AS_STRIDE + base]);
                acc[t] = __builtin_amdgcn_wmma_f32_16x16x32_f16(
                    false, a, false, b, (short)0, acc[t], false, false);
            }
        }
        float buv = bu[e * H_DIM + nb + m16];
#pragma unroll
        for (int t = 0; t < NTILES; ++t) {
#pragma unroll
            for (int g = 0; g < 8; ++g) {
                int m = 16 * t + g + 8 * hi;
                float v = (acc[t][g] + buv) * gwt[m];
                unsafeAtomicAdd(&out[(size_t)toks[m] * H_DIM + nb + m16], v);
            }
        }
    }
}

// ---------------------------------------------------------------- launch
extern "C" void kernel_launch(void* const* d_in, const int* in_sizes, int n_in,
                              void* d_out, int out_size, void* d_ws, size_t ws_size,
                              hipStream_t stream) {
    const float* x  = (const float*)d_in[0];
    const float* Wr = (const float*)d_in[1];
    const float* Wg = (const float*)d_in[2];
    const float* bg = (const float*)d_in[3];
    const float* Wu = (const float*)d_in[4];
    const float* bu = (const float*)d_in[5];
    float* out = (float*)d_out;

    char* ws = (char*)d_ws;
    size_t off = 0;
    int* counts = (int*)(ws + off);            off += 256;
    int* lists  = (int*)(ws + off);            off += (size_t)NEXP * T_TOK * 4;
    float* gw   = (float*)(ws + off);          off += (size_t)NEXP * T_TOK * 4;
    _Float16* pg = (_Float16*)(ws + off);
    off += (size_t)NEXP * FRAGS_PER_EXPERT * FRAG_HALVES * 2;   // 4 MB
    _Float16* pu = (_Float16*)(ws + off);
    off += (size_t)NEXP * FRAGS_PER_EXPERT * FRAG_HALVES * 2;   // 4 MB

    int n4 = (T_TOK * H_DIM) / 4;
    moe_zero_kernel<<<(n4 + 255) / 256, 256, 0, stream>>>(out, counts, n4);
    dim3 pgrid(NEXP * FRAGS_PER_EXPERT, 2);
    moe_pack_kernel<<<pgrid, 256, 0, stream>>>(Wg, Wu, pg, pu);
    moe_router_kernel<<<T_TOK / 8, 256, 0, stream>>>(x, Wr, counts, lists, gw);
    dim3 egrid(MAX_TILES, NEXP);
    moe_expert_kernel<<<egrid, 256, 0, stream>>>(x, pg, bg, pu, bu,
                                                 counts, lists, gw, out);
}
